// DDCMBlock_3152505995485
// MI455X (gfx1250) — matrix operand order repaired and checked by reading, not compiled, not run on hardware
//
#include <hip/hip_runtime.h>

typedef __attribute__((ext_vector_type(16))) __bf16 v16bf;
typedef __attribute__((ext_vector_type(8)))  __bf16 v8bf;
typedef __attribute__((ext_vector_type(8)))  float  v8f;

#define CCH 128

// ---------------- conversion kernels ----------------

__global__ __launch_bounds__(256) void cvt_feat_kernel(const float* __restrict__ f,
                                                       unsigned short* __restrict__ fb,
                                                       int total /* (N+1)*C */,
                                                       int nvalid /* N*C */) {
    int d = blockIdx.x * 256 + threadIdx.x;
    if (d >= total) return;
    float v = (d < nvalid) ? f[d] : 0.0f;   // sentinel row N = zeros
    ((__bf16*)fb)[d] = (__bf16)v;
}

// Pre-swizzle W[3][3][128][128] (== Wcat[3][384][128]) into WMMA B-fragment order:
// flat = (((a*12 + s)*8 + c)*32 + lane)*16 + e
// lane<16 : column = 16c+lane,      K = 32s + e        (K 0..15)
// lane>=16: column = 16c+(lane-16), K = 32s + 16 + e   (K 16..31)
__global__ __launch_bounds__(256) void cvt_w_kernel(const float* __restrict__ W,
                                                    unsigned short* __restrict__ wf) {
    int d = blockIdx.x * 256 + threadIdx.x;
    if (d >= 3 * 12 * 8 * 32 * 16) return;
    int a = d / 49152;
    int r = d - a * 49152;
    int s = r >> 12;            // kstep 0..11
    r &= 4095;
    int c = r >> 9;             // column tile 0..7
    int lane = (r >> 4) & 31;
    int e = r & 15;
    int K = ((lane < 16) ? 0 : 16) + e;
    int kg = 32 * s + K;                      // 0..383 within Wcat
    int j = 16 * c + (lane & 15);             // output channel
    ((__bf16*)wf)[d] = (__bf16)W[((size_t)a * 384 + kg) * CCH + j];
}

// ---------------- GEMM + partial BN statistics ----------------

__global__ __launch_bounds__(256) void gemm_kernel(const unsigned short* __restrict__ featb_,
                                                   const unsigned short* __restrict__ wfrag_,
                                                   const int* __restrict__ nb,
                                                   float* __restrict__ outw,
                                                   float* __restrict__ part,
                                                   int axis, int Nrows, int ntiles, int G) {
    const __bf16* feat  = (const __bf16*)featb_;
    const __bf16* wfrag = (const __bf16*)wfrag_;
    int lane = threadIdx.x & 31;
    int wave = threadIdx.x >> 5;     // column tile owned by this wave (0..7)
    int g    = blockIdx.x;
    int lh   = lane >> 4;            // lane half
    int ln   = lane & 15;

    // Load this wave's 12 B fragments once; they live in VGPRs for the whole kernel.
    v16bf B[12];
#pragma unroll
    for (int s = 0; s < 12; ++s) {
        const __bf16* p = wfrag + ((((size_t)axis * 12 + s) * 8 + wave) * 32 + lane) * 16;
        B[s] = *(const v16bf*)p;
    }

    const int* nbp = nb + ((size_t)axis * 2 + 0) * (size_t)Nrows;
    const int* nbn = nb + ((size_t)axis * 2 + 1) * (size_t)Nrows;

    float csum = 0.f, csq = 0.f;     // per-lane partial column sums (cols of this ctile)

    for (int tile = g; tile < ntiles; tile += G) {
        int rowbase = tile * 16;
        int i  = rowbase + ln;
        bool ok = (i < Nrows);
        int ii = ok ? i : (Nrows - 1);           // clamped, branch-free index load
        int rp = nbp[ii];
        int rn = nbn[ii];
        int ic = ok ? i : Nrows;                 // sentinel zero row for padding rows
        rp = ok ? rp : Nrows;
        rn = ok ? rn : Nrows;

        // Stage all 12 A fragments in independent registers.
        v16bf A[12];
#pragma unroll
        for (int s = 0; s < 12; ++s) {
            int t  = s >> 2;                               // tap: 0=prev,1=self,2=next
            int kb = ((s & 3) << 5) + (lh << 4);           // 32*(s%4) + 16*lanehalf
            int src = (t == 0) ? rp : ((t == 1) ? ic : rn);
            const __bf16* p = feat + (size_t)src * CCH + kb;
            v8bf a0 = *(const v8bf*)p;                     // K lo..lo+7
            v8bf a1 = *(const v8bf*)(p + 8);               // K lo+8..lo+15
            A[s] = __builtin_shufflevector(a0, a1, 0,1,2,3,4,5,6,7,
                                                   8,9,10,11,12,13,14,15);
        }

        v8f acc = {};
#pragma unroll
        for (int s = 0; s < 12; ++s) {
            acc = __builtin_amdgcn_wmma_f32_16x16x32_bf16(false, A[s], false, B[s],
                                                          (short)0, acc, false, false);
        }

#if __has_builtin(__builtin_amdgcn_sched_group_barrier)
        // Pipeline hint: issue all VMEM reads (2 index + 24 A-fragment b128 loads)
        // as one deep in-flight group, then the 12 WMMAs (staggered loadcnt waits).
        __builtin_amdgcn_sched_group_barrier(0x0020, 26, 0);  // VMEM reads
        __builtin_amdgcn_sched_group_barrier(0x0008, 12, 0);  // MFMA/WMMA
#endif

        // per-column sum / sum-of-squares for this tile
        float s1 = 0.f, s2 = 0.f;
#pragma unroll
        for (int v = 0; v < 8; ++v) {
            float x = acc[v];
            s1 += x;
            s2 += x * x;
        }

        // store tile: wave-uniform guard -> unguarded 8-store clause on the fast path
        float* op = outw + ((size_t)rowbase + (lh << 3)) * CCH + wave * 16 + ln;
        if (rowbase + 16 <= Nrows) {
#pragma unroll
            for (int v = 0; v < 8; ++v) op[(size_t)v * CCH] = acc[v];
        } else {
#pragma unroll
            for (int v = 0; v < 8; ++v) {
                int row = rowbase + (lh << 3) + v;
                if (row < Nrows) op[(size_t)v * CCH] = acc[v];
            }
        }

        // fold the two lane halves (rows 0-7 + rows 8-15) with SWAPX16 swizzle
        s1 += __int_as_float(__builtin_amdgcn_ds_swizzle(__float_as_int(s1), 0x401F));
        s2 += __int_as_float(__builtin_amdgcn_ds_swizzle(__float_as_int(s2), 0x401F));
        csum += s1;
        csq  += s2;
    }

    if (lh == 0) {   // lanes 0..15 hold full sums for columns wave*16+ln
        int j = wave * 16 + ln;
        part[((size_t)g * 2 + 0) * CCH + j] = csum;
        part[((size_t)g * 2 + 1) * CCH + j] = csq;
    }
}

// ---------------- BN statistics reduction (deterministic) ----------------

__global__ __launch_bounds__(128) void stats_kernel(const float* __restrict__ part,
                                                    const float* __restrict__ gamma,
                                                    const float* __restrict__ beta,
                                                    float* __restrict__ stats,
                                                    int G, int axis, int Nrows) {
    int j = threadIdx.x;
    float sum = 0.f, sq = 0.f;
    for (int g = 0; g < G; ++g) {
        sum += part[((size_t)g * 2 + 0) * CCH + j];
        sq  += part[((size_t)g * 2 + 1) * CCH + j];
    }
    float invN = 1.0f / (float)Nrows;
    float mu  = sum * invN;
    float var = sq * invN - mu * mu;
    float rs  = rsqrtf(var + 1e-5f);
    float sc  = rs * gamma[axis * CCH + j];
    float sh  = beta[axis * CCH + j] - mu * sc;
    stats[j]        = sc;
    stats[CCH + j]  = sh;
}

// ---------------- BN apply + sigmoid + accumulate (+ final multiply) ----------------

__global__ __launch_bounds__(256) void apply_kernel(const float* __restrict__ outw,
                                                    const float* __restrict__ stats,
                                                    const float* __restrict__ features,
                                                    float* __restrict__ acc,
                                                    int mode, int total4) {
    int q = blockIdx.x * 256 + threadIdx.x;
    if (q >= total4) return;
    int e = q * 4;
    int j = e & (CCH - 1);
    float4 v  = *(const float4*)(outw + e);
    float4 sc = *(const float4*)(stats + j);
    float4 sh = *(const float4*)(stats + CCH + j);
    float4 r;
    r.x = 1.0f / (1.0f + __expf(-(v.x * sc.x + sh.x)));
    r.y = 1.0f / (1.0f + __expf(-(v.y * sc.y + sh.y)));
    r.z = 1.0f / (1.0f + __expf(-(v.z * sc.z + sh.z)));
    r.w = 1.0f / (1.0f + __expf(-(v.w * sc.w + sh.w)));
    if (mode == 0) {
        *(float4*)(acc + e) = r;
    } else if (mode == 1) {
        float4 a = *(const float4*)(acc + e);
        a.x += r.x; a.y += r.y; a.z += r.z; a.w += r.w;
        *(float4*)(acc + e) = a;
    } else {
        float4 a = *(const float4*)(acc + e);
        float4 f = *(const float4*)(features + e);
        a.x = (a.x + r.x) * f.x; a.y = (a.y + r.y) * f.y;
        a.z = (a.z + r.z) * f.z; a.w = (a.w + r.w) * f.w;
        *(float4*)(acc + e) = a;
    }
}

// ---------------- launcher ----------------

extern "C" void kernel_launch(void* const* d_in, const int* in_sizes, int n_in,
                              void* d_out, int out_size, void* d_ws, size_t ws_size,
                              hipStream_t stream) {
    const float* features = (const float*)d_in[0];
    const int*   nb       = (const int*)d_in[1];
    const float* W        = (const float*)d_in[2];
    const float* gamma    = (const float*)d_in[3];
    const float* beta     = (const float*)d_in[4];

    const int N      = in_sizes[0] / CCH;
    const int ntiles = (N + 15) / 16;
    const int G      = 1250;                 // workgroups for the GEMM grid

    // workspace carve-up
    char*  ws  = (char*)d_ws;
    size_t off = 0;
    unsigned short* featb = (unsigned short*)(ws + off);
    off += (size_t)(N + 1) * CCH * sizeof(unsigned short);
    off = (off + 255) & ~(size_t)255;
    unsigned short* wfrag = (unsigned short*)(ws + off);
    off += (size_t)3 * 12 * 8 * 32 * 16 * sizeof(unsigned short);
    off = (off + 255) & ~(size_t)255;
    float* outw = (float*)(ws + off);
    off += (size_t)N * CCH * sizeof(float);
    off = (off + 255) & ~(size_t)255;
    float* part = (float*)(ws + off);
    off += (size_t)G * 2 * CCH * sizeof(float);
    off = (off + 255) & ~(size_t)255;
    float* stats = (float*)(ws + off);
    (void)ws_size; (void)n_in; (void)out_size;

    // 1. precision conversion + weight fragment swizzle
    int ftot = (N + 1) * CCH;
    cvt_feat_kernel<<<(ftot + 255) / 256, 256, 0, stream>>>(features, featb, ftot, N * CCH);
    cvt_w_kernel<<<(3 * 12 * 8 * 32 * 16 + 255) / 256, 256, 0, stream>>>(W, wfrag);

    // 2. per-axis: GEMM -> stats -> apply
    const int total4 = (N * CCH) / 4;
    for (int a = 0; a < 3; ++a) {
        gemm_kernel<<<G, 256, 0, stream>>>(featb, wfrag, nb, outw, part, a, N, ntiles, G);
        stats_kernel<<<1, 128, 0, stream>>>(part, gamma, beta, stats, G, a, N);
        int mode = (a == 0) ? 0 : ((a == 2) ? 2 : 1);
        apply_kernel<<<(total4 + 255) / 256, 256, 0, stream>>>(outw, stats, features,
                                                               (float*)d_out, mode, total4);
    }
}
